// MultiHeadAttention_16750372454929
// MI455X (gfx1250) — matrix operand order repaired and checked by reading, not compiled, not run on hardware
//
#include <hip/hip_runtime.h>
#include <stdint.h>

// ---------------------------------------------------------------------------
// MI455X (gfx1250) implementation of the nonstandard multi-head attention.
// All heavy math runs on v_wmma_f32_16x16x32_bf16 (wave32 WMMA); GEMM tile
// staging uses the CDNA5 async data mover (global_load_async_to_lds_b128,
// tracked by ASYNCcnt).
//
// Workspace layout (bytes, total ~672 MB):
//   [0,128M)    xbf   : x converted to bf16            [32768][2048]
//   [128M,136M) WqT   : Wq^T bf16                      [2048][2048]
//   [136M,144M) WkT   : Wk^T bf16
//   [144M,152M) WvT   : Wv^T bf16
//   [152M,160M) WoT   : Wo^T bf16
//   [160M,288M) qbuf  : Q, layout [b*16+h][i][t]  bf16
//   [288M,416M) kbuf  : K, layout [b*16+h][j][t]  bf16 (transposed vs ref)
//   [416M,544M) vbuf  : V, layout [b*16+h][t][j]  bf16
//   [544M,672M) y2    : pre-Wo activations in the reference's reshape
//                       layout [b][4096][2048] bf16
// ---------------------------------------------------------------------------

typedef __attribute__((ext_vector_type(16))) __bf16 v16bf;
typedef __attribute__((ext_vector_type(8)))  float  v8f;

#define NHEADS 16
#define HDIM   128
#define DMODEL 2048
#define BB     8
#define TT     4096
#define MROWS  (BB * TT)   // 32768

union FragU { v16bf v; uint4 q[2]; };

__device__ __forceinline__ uint16_t f2bf(float f) {
  union { float f; uint32_t u; } x; x.f = f;
  uint32_t u = x.u + 0x7FFFu + ((x.u >> 16) & 1u);   // round-to-nearest-even
  return (uint16_t)(u >> 16);
}

__device__ __forceinline__ v8f zero8() {
  v8f z = {0.f, 0.f, 0.f, 0.f, 0.f, 0.f, 0.f, 0.f};
  return z;
}

__device__ __forceinline__ v8f wmma_bf16(v16bf a, v16bf b, v8f c) {
  return __builtin_amdgcn_wmma_f32_16x16x32_bf16(
      /*neg_a=*/false, a, /*neg_b=*/false, b,
      /*c_mod=*/(short)0, c, /*reuse_a=*/false, /*reuse_b=*/false);
}

// Async copy of 16B global -> LDS (per lane).  INST_OFFSET is added to both
// the LDS and global addresses (ISA 08_async_tensor 4.4), so chunked copies
// share one LDS-address VGPR and one global-address VGPR pair.
__device__ __forceinline__ void async_cp16(uint32_t lds, uint64_t gaddr) {
  asm volatile("global_load_async_to_lds_b128 %0, %1, off" :: "v"(lds), "v"(gaddr) : "memory");
}
__device__ __forceinline__ void async_cp16_off16(uint32_t lds, uint64_t gaddr) {
  asm volatile("global_load_async_to_lds_b128 %0, %1, off offset:16" :: "v"(lds), "v"(gaddr) : "memory");
}
__device__ __forceinline__ void async_cp16_off32(uint32_t lds, uint64_t gaddr) {
  asm volatile("global_load_async_to_lds_b128 %0, %1, off offset:32" :: "v"(lds), "v"(gaddr) : "memory");
}
__device__ __forceinline__ void async_cp16_off48(uint32_t lds, uint64_t gaddr) {
  asm volatile("global_load_async_to_lds_b128 %0, %1, off offset:48" :: "v"(lds), "v"(gaddr) : "memory");
}
__device__ __forceinline__ void wait_async0() {
  asm volatile("s_wait_asynccnt 0x0" ::: "memory");
}
// Low 32 bits of a generic pointer to __shared__ are the LDS byte address
// (ISA flat addressing).
__device__ __forceinline__ uint32_t lds_addr(const void* p) {
  return (uint32_t)(uintptr_t)p;
}

// A fragment (16x32 bf16, M x K): lane L<16 row M=L holds K={kk..kk+7, kk+16..kk+23};
// lane L>=16 row M=L-16 holds K={kk+8..kk+15, kk+24..kk+31}.  rowbase = &A[m*ldk].
__device__ __forceinline__ v16bf load_a_frag(const uint16_t* rowbase, int kk, int lane) {
  FragU f;
  int k0 = kk + ((lane & 16) ? 8 : 0);
  f.q[0] = *reinterpret_cast<const uint4*>(rowbase + k0);
  f.q[1] = *reinterpret_cast<const uint4*>(rowbase + k0 + 16);
  return f.v;
}

// B fragment (32x16 bf16, K x N) read from B^T [N][K]: lane<16 holds K={kk..kk+7, kk+16..kk+23},
// lane>=16 holds K={kk+8..kk+15, kk+24..kk+31}; N = n0 + lane%16.  colbase = &BT[n*ldk].
__device__ __forceinline__ v16bf load_b_frag(const uint16_t* colbase, int kk, int lane) {
  FragU f;
  int k0 = kk + ((lane & 16) ? 8 : 0);
  f.q[0] = *reinterpret_cast<const uint4*>(colbase + k0);
  f.q[1] = *reinterpret_cast<const uint4*>(colbase + k0 + 16);
  return f.v;
}

// ---------------------------------------------------------------------------
// fp32 -> bf16 bulk convert (x)
// ---------------------------------------------------------------------------
__global__ __launch_bounds__(256)
void cvt_f32_bf16(const float* __restrict__ in, uint16_t* __restrict__ out, size_t n) {
  size_t i = ((size_t)blockIdx.x * blockDim.x + threadIdx.x) * 4;
  if (i + 3 < n) {
    float4 f = *reinterpret_cast<const float4*>(in + i);
    ushort4 o;
    o.x = f2bf(f.x); o.y = f2bf(f.y); o.z = f2bf(f.z); o.w = f2bf(f.w);
    *reinterpret_cast<ushort4*>(out + i) = o;
  }
}

// ---------------------------------------------------------------------------
// W [rows][cols] fp32  ->  WT [cols][rows] bf16   (LDS tile transpose)
// ---------------------------------------------------------------------------
__global__ __launch_bounds__(256)
void transpose_to_bf16(const float* __restrict__ W, uint16_t* __restrict__ WT,
                       int rows, int cols) {
  __shared__ float tile[32][33];
  const int bx = blockIdx.x * 32, by = blockIdx.y * 32;
  const int tx = threadIdx.x & 31, ty = threadIdx.x >> 5;
  for (int r = ty; r < 32; r += 8)
    tile[r][tx] = W[(size_t)(by + r) * cols + (bx + tx)];
  __syncthreads();
  for (int r = ty; r < 32; r += 8)
    WT[(size_t)(bx + r) * rows + (by + tx)] = f2bf(tile[tx][r]);
}

// ---------------------------------------------------------------------------
// bf16 WMMA GEMM: C = A[M][2048] * BT[2048][2048]^T + bias.
// Block: 256 threads (8 waves) -> 256x128 C tile; wave tile 64x64
// (4x4 fragments -> 16 WMMAs per k-step of 32).
//
// LDS double-buffered pipeline using the CDNA5 async data mover: the
// global_load_async_to_lds_b128 copies for k-step s+1 are issued before the
// 16 WMMAs of k-step s consume fragments from LDS (ds_load_b128); the only
// synchronization is one s_wait_asynccnt 0 + __syncthreads per k-step, a
// full k-step after issue.  No staging VGPRs, no loadcnt waits.
// Row stride 40 elems (80 B): 16B-aligned and conflict-free for 16-row gathers.
//
// LAYOUT 0: bf16 out, "qstyle"  [b*16+h][c/16][t]
// LAYOUT 1: bf16 out, "kstyle"  [b*16+h][t][c/16]
// LAYOUT 2: fp32 out, row-major [m][n]
// ---------------------------------------------------------------------------
template <int LAYOUT>
__global__ __launch_bounds__(256, 1)
void gemm_bf16_wmma(const uint16_t* __restrict__ A,
                    const uint16_t* __restrict__ BT,
                    const float* __restrict__ bias,
                    void* __restrict__ Cout) {
  constexpr int K = DMODEL;
  constexpr int N = DMODEL;
  constexpr int LDT = 40;                         // LDS row stride (elems)
  __shared__ uint16_t Atile[2][256 * LDT];        // 2 x 20 KB
  __shared__ uint16_t Btile[2][128 * LDT];        // 2 x 10 KB

  const int tid  = threadIdx.x;
  const int lane = tid & 31;
  const int wave = tid >> 5;
  const int nlane = lane & 15;
  const int wm = (wave & 3) * 64;
  const int wn = (wave >> 2) * 64;
  const int M0 = blockIdx.x * 256;
  const int N0 = blockIdx.y * 128;

  v8f acc[4][4];
#pragma unroll
  for (int i = 0; i < 4; i++)
#pragma unroll
    for (int j = 0; j < 4; j++) acc[i][j] = zero8();

  // Staging ownership: thread tid copies A row (M0+tid), 4x16B chunks, and
  // B row (N0+tid/2), 2x16B chunks at elem offset (tid&1)*16.
  const uint16_t* ag = A + (size_t)(M0 + tid) * K;
  const int bn = tid >> 1;
  const int bc = (tid & 1) * 16;
  const uint16_t* bg = BT + (size_t)(N0 + bn) * K + bc;

  const uint32_t laA[2] = { lds_addr(&Atile[0][tid * LDT]),
                            lds_addr(&Atile[1][tid * LDT]) };
  const uint32_t laB[2] = { lds_addr(&Btile[0][bn * LDT + bc]),
                            lds_addr(&Btile[1][bn * LDT + bc]) };

  auto stage_async = [&](int kk, int buf) {
    const uint64_t ga = (uint64_t)(uintptr_t)(ag + kk);
    async_cp16(laA[buf], ga);
    async_cp16_off16(laA[buf], ga);
    async_cp16_off32(laA[buf], ga);
    async_cp16_off48(laA[buf], ga);
    const uint64_t gb = (uint64_t)(uintptr_t)(bg + kk);
    async_cp16(laB[buf], gb);
    async_cp16_off16(laB[buf], gb);
  };

  const int ka0 = (lane & 16) ? 8 : 0;
  const int kb0 = (lane & 16) ? 8 : 0;

  auto compute = [&](int buf) {
    FragU a[4], b[4];
#pragma unroll
    for (int i = 0; i < 4; i++) {
      const uint16_t* p = &Atile[buf][(wm + i * 16 + nlane) * LDT + ka0];
      a[i].q[0] = *reinterpret_cast<const uint4*>(p);       // ds_load_b128
      a[i].q[1] = *reinterpret_cast<const uint4*>(p + 16);
    }
#pragma unroll
    for (int j = 0; j < 4; j++) {
      const uint16_t* p = &Btile[buf][(wn + j * 16 + nlane) * LDT + kb0];
      b[j].q[0] = *reinterpret_cast<const uint4*>(p);
      b[j].q[1] = *reinterpret_cast<const uint4*>(p + 16);
    }
#pragma unroll
    for (int i = 0; i < 4; i++)
#pragma unroll
      for (int j = 0; j < 4; j++)
        acc[i][j] = wmma_bf16(a[i].v, b[j].v, acc[i][j]);
  };

  // Pipeline: preload k-step 0, then steady state, then tail.
  stage_async(0, 0);
  wait_async0();
  __syncthreads();

  int cur = 0;
  for (int kk = 0; kk < K - 32; kk += 32) {
    stage_async(kk + 32, cur ^ 1);   // async copies in flight during compute
    compute(cur);
    wait_async0();                   // this wave's copies landed in LDS
    __syncthreads();                 // everyone's copies visible
    cur ^= 1;
  }
  compute(cur);

  // C/D layout: acc[i][j][r] -> row M0+wm+i*16+r+8*(lane>>4), col N0+wn+j*16+lane%16
  const int rbase = 8 * (lane >> 4);
#pragma unroll
  for (int i = 0; i < 4; i++) {
#pragma unroll
    for (int j = 0; j < 4; j++) {
      const int ncol = N0 + wn + j * 16 + nlane;
      const float bv = bias[ncol];
#pragma unroll
      for (int r = 0; r < 8; r++) {
        const int mrow = M0 + wm + i * 16 + rbase + r;
        const float val = acc[i][j][r] + bv;
        if (LAYOUT == 2) {
          ((float*)Cout)[(size_t)mrow * N + ncol] = val;
        } else {
          const int b = mrow >> 12;          // / T
          const int t = mrow & (TT - 1);
          const int h = ncol & 15;
          const int c16 = ncol >> 4;
          size_t dst;
          if (LAYOUT == 0)
            dst = (((size_t)(b * 16 + h) * HDIM) + c16) * TT + t;
          else
            dst = (((size_t)(b * 16 + h) * TT) + t) * HDIM + c16;
          ((uint16_t*)Cout)[dst] = f2bf(val);
        }
      }
    }
  }
}

// ---------------------------------------------------------------------------
// Attention for one (b,h): scores = Q[128][T] * K^T  (contraction over T),
// softmax over the 128 key-dim, then y = att[128][128] * V[128][T] written
// into the reference's reshape layout.
// ---------------------------------------------------------------------------
__global__ __launch_bounds__(256, 1)
void attention_wmma(const uint16_t* __restrict__ Q,   // [BH][128][T]
                    const uint16_t* __restrict__ Kq,  // [BH][128][T]
                    const uint16_t* __restrict__ Vk,  // [BH][T][128]
                    uint16_t* __restrict__ Y2) {      // [B][4096][2048]
  __shared__ uint16_t att_bf[128][128];               // 32 KB
  const int bh = blockIdx.x;
  const int b = bh >> 4;
  const int h = bh & 15;
  const int lane = threadIdx.x & 31;
  const int wave = threadIdx.x >> 5;
  const int nlane = lane & 15;

  const uint16_t* Qb = Q + (size_t)bh * HDIM * TT;
  const uint16_t* Kb = Kq + (size_t)bh * HDIM * TT;
  const uint16_t* Vb = Vk + (size_t)bh * TT * HDIM;

  // ---- Phase 1: scores.  Wave w owns rows [16w,16w+16) x all 128 cols.
  v8f sc[8];
#pragma unroll
  for (int j = 0; j < 8; j++) sc[j] = zero8();

  const uint16_t* qrow = Qb + (size_t)(wave * 16 + nlane) * TT;
  const uint16_t* kcol = Kb + (size_t)nlane * TT;

  for (int kk = 0; kk < TT; kk += 32) {
    v16bf a = load_a_frag(qrow, kk, lane);
#pragma unroll
    for (int j = 0; j < 8; j++) {
      v16bf bf = load_b_frag(kcol + (size_t)j * 16 * TT, kk, lane);
      sc[j] = wmma_bf16(a, bf, sc[j]);
    }
  }

  // ---- Phase 2: softmax per row, fully in registers.
  // Row m = 16*wave + r + 8*(lane>>4): spread over one 16-lane half x 8 tiles.
  const float scale = 0.08838834764831845f;  // 1/sqrt(128)
#pragma unroll
  for (int r = 0; r < 8; r++) {
    float v[8];
    float mx = -3.4e38f;
#pragma unroll
    for (int j = 0; j < 8; j++) { v[j] = sc[j][r] * scale; mx = fmaxf(mx, v[j]); }
    for (int m = 1; m < 16; m <<= 1) mx = fmaxf(mx, __shfl_xor(mx, m, 32));
    float sum = 0.f;
#pragma unroll
    for (int j = 0; j < 8; j++) { v[j] = __expf(v[j] - mx); sum += v[j]; }
    for (int m = 1; m < 16; m <<= 1) sum += __shfl_xor(sum, m, 32);
    const float inv = 1.0f / sum;
    const int mrow = wave * 16 + r + 8 * (lane >> 4);
#pragma unroll
    for (int j = 0; j < 8; j++) att_bf[mrow][j * 16 + nlane] = f2bf(v[j] * inv);
  }
  __syncthreads();

  // ---- Phase 3: y = att @ V.  Wave w: rows [32*(w&3), +32), t in [(w>>2)*2048, +2048).
  const int wm = (wave & 3) * 32;
  const int thalf = (wave >> 2) * 2048;
  const uint16_t* att0 = &att_bf[wm + nlane][0];
  const uint16_t* att1 = &att_bf[wm + 16 + nlane][0];
  const int rbase = 8 * (lane >> 4);

  for (int tn = 0; tn < 2048; tn += 64) {
    v8f yac[2][4];
#pragma unroll
    for (int i = 0; i < 2; i++)
#pragma unroll
      for (int j = 0; j < 4; j++) yac[i][j] = zero8();

#pragma unroll
    for (int kk = 0; kk < 128; kk += 32) {
      FragU fa0, fa1;
      const int k0 = kk + ((lane & 16) ? 8 : 0);
      fa0.q[0] = *reinterpret_cast<const uint4*>(att0 + k0);        // ds_load_b128
      fa0.q[1] = *reinterpret_cast<const uint4*>(att0 + k0 + 16);
      fa1.q[0] = *reinterpret_cast<const uint4*>(att1 + k0);
      fa1.q[1] = *reinterpret_cast<const uint4*>(att1 + k0 + 16);
#pragma unroll
      for (int j = 0; j < 4; j++) {
        const int t = thalf + tn + j * 16 + nlane;
        v16bf bf = load_b_frag(Vb + (size_t)t * HDIM, kk, lane);
        yac[0][j] = wmma_bf16(fa0.v, bf, yac[0][j]);
        yac[1][j] = wmma_bf16(fa1.v, bf, yac[1][j]);
      }
    }

    // Store into reference reshape: flat (i,h,t) -> row=(i*16+h)*2 + (t>>11), col=t&2047
#pragma unroll
    for (int i = 0; i < 2; i++) {
#pragma unroll
      for (int j = 0; j < 4; j++) {
        const int t = thalf + tn + j * 16 + nlane;
#pragma unroll
        for (int r = 0; r < 8; r++) {
          const int irow = wm + i * 16 + rbase + r;
          const size_t row = (size_t)((irow * 16 + h) * 2 + (t >> 11));
          const size_t dst = ((size_t)b * 4096 + row) * 2048 + (t & 2047);
          Y2[dst] = f2bf(yac[i][j][r]);
        }
      }
    }
  }
}

// ---------------------------------------------------------------------------
extern "C" void kernel_launch(void* const* d_in, const int* in_sizes, int n_in,
                              void* d_out, int out_size, void* d_ws, size_t ws_size,
                              hipStream_t stream) {
  (void)in_sizes; (void)n_in; (void)out_size; (void)ws_size;
  const float* x  = (const float*)d_in[0];
  const float* Wq = (const float*)d_in[1];
  const float* bq = (const float*)d_in[2];
  const float* Wk = (const float*)d_in[3];
  const float* bk = (const float*)d_in[4];
  const float* Wv = (const float*)d_in[5];
  const float* bv = (const float*)d_in[6];
  const float* Wo = (const float*)d_in[7];
  const float* bo = (const float*)d_in[8];

  char* ws = (char*)d_ws;
  const size_t MB = 1ull << 20;
  uint16_t* xbf  = (uint16_t*)(ws);
  uint16_t* WqT  = (uint16_t*)(ws + 128 * MB);
  uint16_t* WkT  = (uint16_t*)(ws + 136 * MB);
  uint16_t* WvT  = (uint16_t*)(ws + 144 * MB);
  uint16_t* WoT  = (uint16_t*)(ws + 152 * MB);
  uint16_t* qbuf = (uint16_t*)(ws + 160 * MB);
  uint16_t* kbuf = (uint16_t*)(ws + 288 * MB);
  uint16_t* vbuf = (uint16_t*)(ws + 416 * MB);
  uint16_t* y2   = (uint16_t*)(ws + 544 * MB);

  const size_t nx = (size_t)MROWS * DMODEL;
  cvt_f32_bf16<<<(unsigned)(nx / 4 / 256), 256, 0, stream>>>(x, xbf, nx);

  dim3 tg(DMODEL / 32, DMODEL / 32);
  transpose_to_bf16<<<tg, 256, 0, stream>>>(Wq, WqT, DMODEL, DMODEL);
  transpose_to_bf16<<<tg, 256, 0, stream>>>(Wk, WkT, DMODEL, DMODEL);
  transpose_to_bf16<<<tg, 256, 0, stream>>>(Wv, WvT, DMODEL, DMODEL);
  transpose_to_bf16<<<tg, 256, 0, stream>>>(Wo, WoT, DMODEL, DMODEL);

  dim3 gg(MROWS / 256, DMODEL / 128);
  gemm_bf16_wmma<0><<<gg, 256, 0, stream>>>(xbf, WqT, bq, qbuf);
  gemm_bf16_wmma<0><<<gg, 256, 0, stream>>>(xbf, WkT, bk, kbuf);
  gemm_bf16_wmma<1><<<gg, 256, 0, stream>>>(xbf, WvT, bv, vbuf);

  attention_wmma<<<BB * NHEADS, 256, 0, stream>>>(qbuf, kbuf, vbuf, y2);

  gemm_bf16_wmma<2><<<gg, 256, 0, stream>>>(y2, WoT, bo, d_out);
}
